// Module_1_32830730011406
// MI455X (gfx1250) — matrix operand order repaired and checked by reading, not compiled, not run on hardware
//
#include <hip/hip_runtime.h>
#include <hip/hip_bf16.h>
#include <math.h>

// ---------------------------------------------------------------------------
// MI455X (gfx1250) FFT channel attention.
// - All heavy stages run through v_wmma_f32_16x16x32_f16.
// - All intermediate complex planes stored as f16 (~0.5 GB HBM traffic,
//   ~22us roofline floor at 23.3 TB/s).
// - DFT cos/sin tables staged into LDS via global_load_async_to_lds_b128.
// - Strided B tiles staged per-wave into LDS with double-buffered async
//   copies (s_wait_asynccnt 4 keeps one batch in flight), consumed as
//   ds_load_u16 with immediate offsets (no 64-bit VALU address math).
// ---------------------------------------------------------------------------

typedef _Float16 half_t;
typedef __attribute__((ext_vector_type(16))) _Float16 v16h;
typedef __attribute__((ext_vector_type(8)))  _Float16 v8h;
typedef __attribute__((ext_vector_type(8)))  float    v8f;

union V16U { v16h v; v8h h[2]; };

__device__ __forceinline__ v8f wmma_step(v16h a, v16h b, v8f c) {
  return __builtin_amdgcn_wmma_f32_16x16x32_f16(false, a, false, b, (short)0, c,
                                                false, false);
}

// 16B async global->LDS copy (per-lane). ISA 15.18.3 op 98, tracked by ASYNCcnt.
__device__ __forceinline__ void async_load16(unsigned lds_off,
                                             unsigned long long gaddr) {
#if defined(__AMDGCN__)
  asm volatile("global_load_async_to_lds_b128 %0, %1, off"
               :: "v"(lds_off), "v"(gaddr) : "memory");
#endif
}
__device__ __forceinline__ void wait_async0() {
#if __has_builtin(__builtin_amdgcn_s_wait_asynccnt)
  __builtin_amdgcn_s_wait_asynccnt(0);
#elif defined(__AMDGCN__)
  asm volatile("s_wait_asynccnt 0" ::: "memory");
#endif
}
__device__ __forceinline__ void wait_async4() {   // allow one staged batch in flight
#if __has_builtin(__builtin_amdgcn_s_wait_asynccnt)
  __builtin_amdgcn_s_wait_asynccnt(4);
#elif defined(__AMDGCN__)
  asm volatile("s_wait_asynccnt 4" ::: "memory");
#endif
}

__device__ __forceinline__ unsigned lds_addr(const void* p) {
  // generic shared pointer: low 32 bits are the LDS byte offset (ISA 10.2)
  return (unsigned)(unsigned long long)p;
}

// ---------------------------------------------------------------------------
// Generic complex GEMM vs a precomputed f16 DFT matrix F (L x L), f16 planes:
//   out[M][N] = scale * sum_K (Fc + i*Fs)[M][K] * (inR + i*inI)[K][N]
// N decomposes as chunk = N/inner, pos = N%inner. KCONTIG selects the
// in_k_stride==1 fast path (direct v8h global loads); otherwise B tiles are
// async-staged into per-wave LDS double buffers (requires in_pos_stride==1,
// 16 | inner -- true for all strided uses here).
// One wave = one 16x16 tile; block = 4 waves along N.
// ---------------------------------------------------------------------------
template <bool KCONTIG>
__global__ void cgemm_dft(const half_t* __restrict__ inR, const half_t* __restrict__ inI,
                          half_t* __restrict__ outR, half_t* __restrict__ outI,
                          const half_t* __restrict__ Fc, const half_t* __restrict__ Fs,
                          int L, long inner,
                          long chunk_stride, long in_pos_stride, long out_pos_stride,
                          long in_k_stride, long out_m_stride, float scale)
{
  __shared__ __align__(16) half_t lF[2 * 9216];          // cos at 0, sin at +9216
  __shared__ __align__(16) half_t lB[KCONTIG ? 16 : 8192]; // 4 waves x 2 bufs x 2 planes x 32x16

  {
    const int chunks = (L * L) >> 3;                      // 16B chunks per table
    const unsigned ldsC = lds_addr(&lF[0]);
    const unsigned ldsS = lds_addr(&lF[9216]);
    for (int i = threadIdx.x; i < chunks; i += (int)blockDim.x) {
      async_load16(ldsC + i * 16,
                   (unsigned long long)(const char*)Fc + (unsigned long long)i * 16);
      async_load16(ldsS + i * 16,
                   (unsigned long long)(const char*)Fs + (unsigned long long)i * 16);
    }
    wait_async0();
    __syncthreads();
  }

  const int lane = threadIdx.x & 31;
  const int wave = threadIdx.x >> 5;
  const int lh = lane & 15;            // B/D column, A row offset
  const int hi = lane >> 4;            // K-half / M-half selector
  const long ntile = (long)blockIdx.x * 4 + wave;
  const long Ng0 = ntile * 16;         // first column of this wave's tile
  const long ch  = Ng0 / inner;        // wave-uniform (16 | inner)
  const long pos0 = Ng0 % inner;
  const int  m0  = blockIdx.y * 16;

  const half_t* bR = inR + ch * chunk_stride + pos0 * in_pos_stride;
  const half_t* bI = inI + ch * chunk_stride + pos0 * in_pos_stride;
  const half_t* lFc = &lF[0];
  const half_t* lFs = &lF[9216];
  half_t* myB = &lB[0] + wave * 2048;  // 2 bufs * (2 planes * 512 halfs)

  // Stage one 32x16 B tile (both planes) for K rows [k0, k0+32): row per lane,
  // 32B per row as two async 16B copies. 4 async instructions per wave.
  auto stage = [&](int buf, int k0) {
    const unsigned l0 = lds_addr(myB + buf * 1024 + lane * 16);
    const unsigned long long gR =
        (unsigned long long)(bR + (long)(k0 + lane) * in_k_stride);
    const unsigned long long gI =
        (unsigned long long)(bI + (long)(k0 + lane) * in_k_stride);
    async_load16(l0,             gR);
    async_load16(l0 + 16,        gR + 16);
    async_load16(l0 + 1024,      gI);        // plane I at +512 halfs
    async_load16(l0 + 1024 + 16, gI + 16);
  };

  v8f accR = {}; v8f accI = {};
  if (!KCONTIG) stage(0, 0);

  for (int k0 = 0; k0 < L; k0 += 32) {
    V16U ac, asv, ansv;
    const v8h* cr = (const v8h*)(lFc + (m0 + lh) * L + k0 + hi * 16);
    const v8h* sr = (const v8h*)(lFs + (m0 + lh) * L + k0 + hi * 16);
    ac.h[0]  = cr[0]; ac.h[1]  = cr[1];      // +8 halfs = +1 v8h (ds_load_b128)
    asv.h[0] = sr[0]; asv.h[1] = sr[1];
    #pragma unroll
    for (int j = 0; j < 16; ++j) ansv.v[j] = -asv.v[j];

    v16h br, bi;
    if (KCONTIG) {
      const v8h* pr = (const v8h*)(bR + k0 + hi * 16);
      const v8h* pi = (const v8h*)(bI + k0 + hi * 16);
      V16U tr, ti;
      tr.h[0] = pr[0]; tr.h[1] = pr[1];
      ti.h[0] = pi[0]; ti.h[1] = pi[1];
      br = tr.v; bi = ti.v;
    } else {
      const int buf = (k0 >> 5) & 1;
      if (k0 + 32 < L) { stage(buf ^ 1, k0 + 32); wait_async4(); }
      else             { wait_async0(); }
      const half_t* bb = myB + buf * 1024;   // [plane][k(32)][n(16)]
      #pragma unroll
      for (int j = 0; j < 16; ++j) {
        br[j] = bb[(j + hi * 16) * 16 + lh];        // ds_load_u16, imm offsets
        bi[j] = bb[512 + (j + hi * 16) * 16 + lh];
      }
    }
    accR = wmma_step(ac.v,   br, accR);   // + Fr*br
    accI = wmma_step(asv.v,  br, accI);   // + Fs*br
    accR = wmma_step(ansv.v, bi, accR);   // - Fs*bi
    accI = wmma_step(ac.v,   bi, accI);   // + Fr*bi
  }

  const long obase = ch * chunk_stride + (pos0 + lh) * out_pos_stride;
  #pragma unroll
  for (int r = 0; r < 8; ++r) {
    const long addr = obase + (long)(m0 + r + hi * 8) * out_m_stride;
    outR[addr] = (_Float16)(accR[r] * scale);
    outI[addr] = (_Float16)(accI[r] * scale);
  }
}

// Real-input variant (imag = 0), contiguous K: fft over w of the conv output.
__global__ void cgemm_dft_real(const half_t* __restrict__ inR,
                               half_t* __restrict__ outR, half_t* __restrict__ outI,
                               const half_t* __restrict__ Fc, const half_t* __restrict__ Fs,
                               int L, long chunk_stride)
{
  __shared__ __align__(16) half_t lF[2 * 9216];
  {
    const int chunks = (L * L) >> 3;
    const unsigned ldsC = lds_addr(&lF[0]);
    const unsigned ldsS = lds_addr(&lF[9216]);
    for (int i = threadIdx.x; i < chunks; i += (int)blockDim.x) {
      async_load16(ldsC + i * 16,
                   (unsigned long long)(const char*)Fc + (unsigned long long)i * 16);
      async_load16(ldsS + i * 16,
                   (unsigned long long)(const char*)Fs + (unsigned long long)i * 16);
    }
    wait_async0();
    __syncthreads();
  }

  const int lane = threadIdx.x & 31, wave = threadIdx.x >> 5;
  const int lh = lane & 15, hi = lane >> 4;
  const long Ng = ((long)blockIdx.x * 4 + wave) * 16 + lh;   // row index (b,c,h)
  const int  m0 = blockIdx.y * 16;
  const half_t* bR = inR + Ng * chunk_stride;
  const half_t* lFc = &lF[0];
  const half_t* lFs = &lF[9216];

  v8f accR = {}; v8f accI = {};
  for (int k0 = 0; k0 < L; k0 += 32) {
    V16U ac, asv;
    const v8h* cr = (const v8h*)(lFc + (m0 + lh) * L + k0 + hi * 16);
    const v8h* sr = (const v8h*)(lFs + (m0 + lh) * L + k0 + hi * 16);
    ac.h[0]  = cr[0]; ac.h[1]  = cr[1];
    asv.h[0] = sr[0]; asv.h[1] = sr[1];

    const v8h* pr = (const v8h*)(bR + k0 + hi * 16);        // contiguous K
    V16U tr; tr.h[0] = pr[0]; tr.h[1] = pr[1];

    accR = wmma_step(ac.v,  tr.v, accR);
    accI = wmma_step(asv.v, tr.v, accI);
  }
  const long obase = Ng * chunk_stride;
  #pragma unroll
  for (int r = 0; r < 8; ++r) {
    const long addr = obase + (long)(m0 + r + hi * 8);
    outR[addr] = (_Float16)accR[r];
    outI[addr] = (_Float16)accI[r];
  }
}

// ---------------------------------------------------------------------------
// Real GEMM for the 1x1 convs: Y[b][o][n] = sum_c W[o][c] * X[b][c][n] (+bias)
// TI: f32 (network input x) or f16 (abs plane); TO: f16 (plane) or f32 (d_out)
// ---------------------------------------------------------------------------
template <typename TI, typename TO>
__global__ void gemm256(const half_t* __restrict__ W, const TI* __restrict__ X,
                        const float* __restrict__ bias, TO* __restrict__ Y,
                        long nc /* 9216 */)
{
  const int lane = threadIdx.x & 31, wave = threadIdx.x >> 5;
  const int lh = lane & 15, hi = lane >> 4;
  const long batch = blockIdx.z;
  const long n0 = ((long)blockIdx.x * 4 + wave) * 16;
  const int  m0 = blockIdx.y * 16;

  const TI* xb = X + batch * 256 * nc + n0 + lh;
  TO*       yb = Y + batch * 256 * nc + n0 + lh;

  v8f acc = {};
  for (int k0 = 0; k0 < 256; k0 += 32) {
    V16U a;
    const v8h* ar = (const v8h*)(W + (long)(m0 + lh) * 256 + k0 + hi * 16);
    a.h[0] = ar[0]; a.h[1] = ar[1];

    v16h b;
    const TI* p = xb + (long)(k0 + hi * 16) * nc;
    #pragma unroll
    for (int j = 0; j < 16; ++j) b[j] = (_Float16)p[(long)j * nc];
    if (k0 < 224) __builtin_prefetch(p + 32 * nc, 0, 1);   // global_prefetch_b8
    acc = wmma_step(a.v, b, acc);
  }
  #pragma unroll
  for (int r = 0; r < 8; ++r) {
    const int mrow = m0 + r + hi * 8;
    float v = acc[r];
    if (bias) v += bias[mrow];
    yb[(long)mrow * nc] = (TO)v;
  }
}

// ---------------------------------------------------------------------------
// attn[b,h] = temperature[h] * q . k^T  (complex 32x32, K = hw = 9216, f16 in)
// Fully vectorized: A and B fragments are raw v8h pairs, no conversions.
// ---------------------------------------------------------------------------
__global__ void attn_qk(const half_t* __restrict__ QR, const half_t* __restrict__ QI,
                        const half_t* __restrict__ KR, const half_t* __restrict__ KI,
                        const float* __restrict__ temp,
                        float* __restrict__ AR, float* __restrict__ AI)
{
  const int lane = threadIdx.x & 31, wave = threadIdx.x >> 5;
  const int lh = lane & 15, hi = lane >> 4;
  const int bh = blockIdx.x, head = bh & 7;
  const int m0 = (wave >> 1) * 16, n0 = (wave & 1) * 16;
  const long HW = 9216;

  const half_t* qr = QR + (long)(bh * 32 + m0 + lh) * HW;
  const half_t* qi = QI + (long)(bh * 32 + m0 + lh) * HW;
  const half_t* kr = KR + (long)(bh * 32 + n0 + lh) * HW;
  const half_t* ki = KI + (long)(bh * 32 + n0 + lh) * HW;

  v8f accR = {}, accI = {};
  for (int k0 = 0; k0 < 9216; k0 += 32) {
    V16U aqr, aqi, naqi, bkr, bki;
    const v8h* pqr = (const v8h*)(qr + k0 + hi * 16);
    const v8h* pqi = (const v8h*)(qi + k0 + hi * 16);
    aqr.h[0] = pqr[0]; aqr.h[1] = pqr[1];     // A layout: 16 contiguous K
    aqi.h[0] = pqi[0]; aqi.h[1] = pqi[1];
    #pragma unroll
    for (int j = 0; j < 16; ++j) naqi.v[j] = -aqi.v[j];

    const v8h* pkr = (const v8h*)(kr + k0 + hi * 16);
    const v8h* pki = (const v8h*)(ki + k0 + hi * 16);
    bkr.h[0] = pkr[0]; bkr.h[1] = pkr[1];     // B layout: 16 contiguous K
    bki.h[0] = pki[0]; bki.h[1] = pki[1];

    accR = wmma_step(aqr.v,  bkr.v, accR);
    accR = wmma_step(naqi.v, bki.v, accR);
    accI = wmma_step(aqr.v,  bki.v, accI);
    accI = wmma_step(aqi.v,  bkr.v, accI);
  }
  const float t = temp[head];
  #pragma unroll
  for (int r = 0; r < 8; ++r) {
    const long addr = (long)bh * 1024 + (long)(m0 + r + hi * 8) * 32 + n0 + lh;
    AR[addr] = accR[r] * t;
    AI[addr] = accI[r] * t;
  }
}

// Separate real/imag softmax over rows of 32 (one wave per row)
__global__ void softmax32(float* __restrict__ AR, float* __restrict__ AI)
{
  const int lane = threadIdx.x & 31, wave = threadIdx.x >> 5;
  const long row = (long)blockIdx.x * 4 + wave;             // 0..4095
  float* base = (row < 2048) ? (AR + row * 32) : (AI + (row - 2048) * 32);
  float v = base[lane];
  float m = v;
  for (int off = 16; off > 0; off >>= 1) m = fmaxf(m, __shfl_xor(m, off, 32));
  const float e = __expf(v - m);
  float s = e;
  for (int off = 16; off > 0; off >>= 1) s += __shfl_xor(s, off, 32);
  base[lane] = e / s;
}

// out[b,h] = attn (32x32 f32) . v (32x9216 f16), complex; one K=32 WMMA step.
// B loads are lane-coalesced (n varies across lanes).
__global__ void attn_v(const float* __restrict__ AR, const float* __restrict__ AI,
                       const half_t* __restrict__ VR, const half_t* __restrict__ VI,
                       half_t* __restrict__ OR_, half_t* __restrict__ OI_)
{
  const int lane = threadIdx.x & 31, wave = threadIdx.x >> 5;
  const int lh = lane & 15, hi = lane >> 4;
  const int bh = blockIdx.z, m0 = blockIdx.y * 16;
  const long n = ((long)blockIdx.x * 4 + wave) * 16 + lh;
  const long HW = 9216;

  v16h ar, ai, nai, bvr, bvi;
  const float* par = AR + (long)bh * 1024 + (long)(m0 + lh) * 32;
  const float* pai = AI + (long)bh * 1024 + (long)(m0 + lh) * 32;
  #pragma unroll
  for (int j = 0; j < 16; ++j) {
    const int kk = j + hi * 16;
    ar[j] = (_Float16)par[kk];
    ai[j] = (_Float16)pai[kk];
    nai[j] = -ai[j];
  }
  const half_t* pvr = VR + (long)bh * 32 * HW + n;
  const half_t* pvi = VI + (long)bh * 32 * HW + n;
  #pragma unroll
  for (int j = 0; j < 16; ++j) {
    const int kk = j + hi * 16;
    bvr[j] = pvr[(long)kk * HW];
    bvi[j] = pvi[(long)kk * HW];
  }
  v8f accR = {}, accI = {};
  accR = wmma_step(ar,  bvr, accR);
  accR = wmma_step(nai, bvi, accR);
  accI = wmma_step(ar,  bvi, accI);
  accI = wmma_step(ai,  bvr, accI);
  #pragma unroll
  for (int r = 0; r < 8; ++r) {
    const long addr = ((long)bh * 32 + m0 + r + hi * 8) * HW + n;
    OR_[addr] = (_Float16)accR[r];
    OI_[addr] = (_Float16)accI[r];
  }
}

// L2 normalize complex f16 rows (length 9216) in place; one block per row
__global__ void l2norm_rows(half_t* __restrict__ re, half_t* __restrict__ im, int rowlen)
{
  __shared__ float red[256];
  const long base = (long)blockIdx.x * rowlen;
  float s = 0.f;
  for (int i = threadIdx.x; i < rowlen; i += 256) {
    const float a = (float)re[base + i], b = (float)im[base + i];
    s += a * a + b * b;
  }
  red[threadIdx.x] = s; __syncthreads();
  for (int off = 128; off > 0; off >>= 1) {
    if (threadIdx.x < off) red[threadIdx.x] += red[threadIdx.x + off];
    __syncthreads();
  }
  const float n  = sqrtf(red[0]);
  const float sc = 1.0f / fmaxf(n, 1e-12f);
  for (int i = threadIdx.x; i < rowlen; i += 256) {
    re[base + i] = (_Float16)((float)re[base + i] * sc);
    im[base + i] = (_Float16)((float)im[base + i] * sc);
  }
}

// Cooley-Tukey inner twiddle for the 9216-pt IDFT: *= e^{+2*pi*i*k1*n2/9216}
__global__ void twiddle9216(half_t* __restrict__ TR, half_t* __restrict__ TI)
{
  const long idx = (long)blockIdx.x * 256 + threadIdx.x;
  const long rem = idx % 9216;
  const int k1 = (int)(rem / 96), n2 = (int)(rem % 96);
  const float ang = 6.28318530717958647692f * (float)((k1 * n2) % 9216) / 9216.0f;
  const float c = __cosf(ang), s = __sinf(ang);
  const float tr = (float)TR[idx], ti = (float)TI[idx];
  TR[idx] = (_Float16)(tr * c - ti * s);
  TI[idx] = (_Float16)(tr * s + ti * c);
}

__global__ void absmag(const half_t* __restrict__ R, const half_t* __restrict__ I,
                       half_t* __restrict__ O)
{
  const long idx = (long)blockIdx.x * 256 + threadIdx.x;
  const float a = (float)R[idx], b = (float)I[idx];
  O[idx] = (_Float16)sqrtf(a * a + b * b);
}

// f16 DFT tables: forward (cos, -sin) / inverse (cos, +sin) L=96; inverse L=32.
__global__ void build_tables(half_t* c96, half_t* s96n, half_t* s96p,
                             half_t* c32, half_t* s32p)
{
  const int idx = blockIdx.x * 256 + threadIdx.x;
  if (idx < 9216) {
    const int m = idx / 96, k = idx % 96;
    const float ang = 6.28318530717958647692f * (float)((m * k) % 96) / 96.0f;
    c96[idx]  = (_Float16)__cosf(ang);
    s96n[idx] = (_Float16)(-__sinf(ang));
    s96p[idx] = (_Float16)__sinf(ang);
  } else if (idx < 9216 + 1024) {
    const int t = idx - 9216, m = t / 32, k = t % 32;
    const float ang = 6.28318530717958647692f * (float)((m * k) % 32) / 32.0f;
    c32[t]  = (_Float16)__cosf(ang);
    s32p[t] = (_Float16)__sinf(ang);
  }
}

__global__ void cvt_weights(const float* __restrict__ w1, const float* __restrict__ w2,
                            const float* __restrict__ w3, const float* __restrict__ wo,
                            half_t* __restrict__ dst)
{
  const int idx = blockIdx.x * 256 + threadIdx.x;     // < 262144
  const int m = idx >> 16, r = idx & 65535;
  const float* src = (m == 0) ? w1 : (m == 1) ? w2 : (m == 2) ? w3 : wo;
  dst[idx] = (_Float16)src[r];
}

// ---------------------------------------------------------------------------
// Orchestration. Workspace (halfs), P = b*c*hw = 18,874,368:
//   Ah | T1R,T1I | QR,QI,KR,KI,VR,VI  (9 f16 planes ~ 340 MB)
//   then f16 weights + DFT tables, then f32 attention planes (~1.1 MB).
// ---------------------------------------------------------------------------
extern "C" void kernel_launch(void* const* d_in, const int* in_sizes, int n_in,
                              void* d_out, int out_size, void* d_ws, size_t ws_size,
                              hipStream_t stream)
{
  const float* x    = (const float*)d_in[0];
  const float* w1   = (const float*)d_in[1];
  const float* b1   = (const float*)d_in[2];
  const float* w2   = (const float*)d_in[3];
  const float* b2   = (const float*)d_in[4];
  const float* w3   = (const float*)d_in[5];
  const float* b3   = (const float*)d_in[6];
  const float* wout = (const float*)d_in[7];
  const float* temp = (const float*)d_in[8];
  float* out = (float*)d_out;

  half_t* hs = (half_t*)d_ws;
  const long P = 18874368L;                 // 8*256*9216
  half_t* Ah  = hs;
  half_t* T1R = hs + 1 * P; half_t* T1I = hs + 2 * P;
  half_t* QR  = hs + 3 * P; half_t* QI  = hs + 4 * P;
  half_t* KR  = hs + 5 * P; half_t* KI  = hs + 6 * P;
  half_t* VR  = hs + 7 * P; half_t* VI  = hs + 8 * P;

  half_t* wh = hs + 9 * P;
  half_t* Wq = wh;            half_t* Wk = wh + 65536;
  half_t* Wv = wh + 131072;   half_t* Wo = wh + 196608;
  half_t* c96  = wh + 262144;
  half_t* s96n = c96 + 9216;
  half_t* s96p = c96 + 18432;
  half_t* c32  = c96 + 27648;
  half_t* s32p = c96 + 28672;
  float* AR = (float*)(wh + 291840);        // 16B aligned
  float* AI = AR + 65536;

  cvt_weights <<<1024, 256, 0, stream>>>(w1, w2, w3, wout, wh);
  build_tables<<<40,   256, 0, stream>>>(c96, s96n, s96p, c32, s32p);

  const half_t* Wt[3] = {Wq, Wk, Wv};
  const float*  bt[3] = {b1, b2, b3};
  half_t*       tR[3] = {QR, KR, VR};
  half_t*       tI[3] = {QI, KI, VI};

  for (int t = 0; t < 3; ++t) {
    // 1x1 conv GEMM -> Ah (f16)
    gemm256<float, half_t><<<dim3(144, 16, 8), 128, 0, stream>>>(Wt[t], x, bt[t], Ah, 9216L);
    // fft over w (rows of 96, real f16 input, contiguous K)
    cgemm_dft_real<<<dim3(3072, 6), 128, 0, stream>>>(Ah, T1R, T1I, c96, s96n, 96, 96L);
    // fft over h: chunk=(b,c), pos=w-freq, K stride 96 (async-staged B)
    cgemm_dft<false><<<dim3(3072, 6), 128, 0, stream>>>(
        T1R, T1I, tR[t], tI[t], c96, s96n, 96,
        96L, 9216L, 1L, 1L, 96L, 96L, 1.0f);
  }

  l2norm_rows<<<2048, 256, 0, stream>>>(QR, QI, 9216);
  l2norm_rows<<<2048, 256, 0, stream>>>(KR, KI, 9216);

  attn_qk  <<<64,   128, 0, stream>>>(QR, QI, KR, KI, temp, AR, AI);
  softmax32<<<1024, 128, 0, stream>>>(AR, AI);
  // out = attn . v -> reuse (QR,QI) as complex output planes
  attn_v<<<dim3(144, 2, 64), 128, 0, stream>>>(AR, AI, VR, VI, QR, QI);

  // ifft over hw = 9216 = 96*96 (Cooley-Tukey):
  // stage 1: inner 96-pt IDFT over n1 (K stride 96, async-staged B)
  cgemm_dft<false><<<dim3(3072, 6), 128, 0, stream>>>(
      QR, QI, T1R, T1I, c96, s96p, 96,
      96L, 9216L, 1L, 1L, 96L, 96L, 1.0f);
  twiddle9216<<<73728, 256, 0, stream>>>(T1R, T1I);
  // stage 2: outer 96-pt IDFT over n2 (contiguous K fast path), 1/9216
  cgemm_dft<true><<<dim3(3072, 6), 128, 0, stream>>>(
      T1R, T1I, KR, KI, c96, s96p, 96,
      96L, 9216L, 96L, 1L, 1L, 96L, 1.0f / 9216.0f);
  // ifft over cp = 32 (single WMMA K-step, K stride 9216, async-staged B)
  cgemm_dft<false><<<dim3(9216, 2), 128, 0, stream>>>(
      KR, KI, QR, QI, c32, s32p, 32,
      9216L, 294912L, 1L, 1L, 9216L, 9216L, 1.0f / 32.0f);

  absmag<<<73728, 256, 0, stream>>>(QR, QI, Ah);

  // final projection (no bias) -> d_out (f32)
  gemm256<half_t, float><<<dim3(144, 16, 8), 128, 0, stream>>>(Wo, Ah, nullptr, out, 9216L);

  (void)in_sizes; (void)n_in; (void)out_size; (void)ws_size;
}
